// GraphGNN_84636625535112
// MI455X (gfx1250) — matrix operand order repaired and checked by reading, not compiled, not run on hardware
//
#include <hip/hip_runtime.h>

typedef __attribute__((ext_vector_type(16))) _Float16 v16h;
typedef __attribute__((ext_vector_type(8)))  float    v8f;
typedef __attribute__((ext_vector_type(4)))  float    v4f;

namespace {
constexpr int kB         = 2;
constexpr int kN         = 50000;
constexpr int kF         = 16;
constexpr int kE         = 1600000;
constexpr int kEH        = 32;
constexpr int kEOUT      = 30;
constexpr int kNOUT      = 16;
constexpr int kAggStride = 32;               // E_OUT padded to 32 for clean b128 loads
constexpr int kEdgeTiles = kE / 16;          // 100000
constexpr int kEdgeBlocks = 625;             // 625 blocks * 8 waves = 5000 waves -> exactly 20 tiles each
constexpr int kNodeTiles = kN / 16;          // 3125
constexpr int kNodeBlocks = (kNodeTiles + 7) / 8;
constexpr int kLdsStride = 36;               // padded to reduce bank conflicts
} // namespace

// Fast sigmoid: v_exp_f32 + v_rcp_f32 (both TRANS, co-execute with VALU).
// 1-ulp rcp instead of correctly-rounded divide: results are truncated to f16
// or are final activations, so this is well within tolerance.
__device__ __forceinline__ float gnn_sigmoid(float x) {
  return __builtin_amdgcn_rcpf(1.0f + __expf(-x));
}

// ---------------------------------------------------------------------------
// Zero the (B, N, 32) aggregation workspace (d_ws is poisoned by the harness).
// ---------------------------------------------------------------------------
__global__ void gnn_zero_kernel(float4* __restrict__ ws, int n4) {
  int i = blockIdx.x * blockDim.x + threadIdx.x;
  const int stride = gridDim.x * blockDim.x;
  for (; i < n4; i += stride) ws[i] = make_float4(0.f, 0.f, 0.f, 0.f);
}

// ---------------------------------------------------------------------------
// Edge MLP (33 -> 32 -> 30, sigmoid) + signed scatter into agg.
// One wave handles a tile of 16 edges via v_wmma_f32_16x16x32_f16.
// ---------------------------------------------------------------------------
__global__ __launch_bounds__(256) void gnn_edge_kernel(
    const float* __restrict__ nf, const float* __restrict__ ew,
    const int* __restrict__ eidx,
    const float* __restrict__ W1, const float* __restrict__ b1v,
    const float* __restrict__ W2, const float* __restrict__ b2v,
    float* __restrict__ agg)
{
  __shared__ float h1s[8][16][kLdsStride];   // per-wave 16x32 activation staging

  const int lane = threadIdx.x & 31;
  const int wid  = threadIdx.x >> 5;
  const int half = lane >> 4;                // 0: lanes 0-15, 1: lanes 16-31
  const int l15  = lane & 15;
  const int koff = half * 16;                // unused; B fragments start at K = 8*half
  const int b    = blockIdx.y;
  const int cb   = 16 + l15;                 // column for the second N-tile

  // --- Build B fragments (ISA 16-bit B layout: lane = column, halves split K) ---
  v16h B1a, B1b, B2a, B2b;
#pragma unroll
  for (int j = 0; j < 16; ++j) {
    const int k = (j < 8) ? 8 * half + j : 16 + 8 * half + (j - 8);
    B1a[j] = (_Float16)W1[k * kEH + l15];
    B1b[j] = (_Float16)W1[k * kEH + 16 + l15];
    B2a[j] = (_Float16)W2[k * kEOUT + l15];
    B2b[j] = (_Float16)((cb < kEOUT) ? W2[k * kEOUT + cb] : 0.0f);
  }
  const float w1ewA = W1[32 * kEH + l15];        // row 32 of W1 = edge-weight row
  const float w1ewB = W1[32 * kEH + 16 + l15];
  const float b1a = b1v[l15];
  const float b1b = b1v[16 + l15];
  const float b2a = b2v[l15];
  const float b2b = (cb < kEOUT) ? b2v[cb] : 0.0f;

  const float* nfb  = nf  + (size_t)b * kN * kF;
  const float* ewb  = ew  + (size_t)b * kE;
  float*       aggb = agg + (size_t)b * kN * kAggStride;
  const int* srcI = eidx;          // edge_index row 0
  const int* tgtI = eidx + kE;     // edge_index row 1

  const int wavesTotal = gridDim.x * 8;
  float (*my)[kLdsStride] = h1s[wid];

  for (int t = blockIdx.x * 8 + wid; t < kEdgeTiles; t += wavesTotal) {
    const int e0 = t * 16;
    const int s0 = srcI[e0 + l15];             // ids for edge row (lane&15)
    const int t0 = tgtI[e0 + l15];

    // Gather features: A layout wants src feats at K 0..15, tgt feats at K 16..31.
    const float* ps = nfb + (size_t)s0 * kF + 8 * half;
    const float* pt = nfb + (size_t)t0 * kF + 8 * half;
    const v4f sLo = *(const v4f*)(ps);
    const v4f sHi = *(const v4f*)(ps + 4);
    const v4f tLo = *(const v4f*)(pt);
    const v4f tHi = *(const v4f*)(pt + 4);

    v16h a1;
#pragma unroll
    for (int j = 0; j < 4; ++j) {
      a1[j]      = (_Float16)sLo[j];
      a1[4 + j]  = (_Float16)sHi[j];
      a1[8 + j]  = (_Float16)tLo[j];
      a1[12 + j] = (_Float16)tHi[j];
    }

    v8f c1a = {};
    v8f c1b = {};
    c1a = __builtin_amdgcn_wmma_f32_16x16x32_f16(false, a1, false, B1a, (short)0, c1a, false, false);
    c1b = __builtin_amdgcn_wmma_f32_16x16x32_f16(false, a1, false, B1b, (short)0, c1b, false, false);

    // Rank-1 edge-weight column (K=32) + bias + sigmoid, staged to LDS row-major.
    const v4f ewLo = *(const v4f*)(ewb + e0 + 8 * half);
    const v4f ewHi = *(const v4f*)(ewb + e0 + 8 * half + 4);
#pragma unroll
    for (int r = 0; r < 8; ++r) {
      const float w = (r < 4) ? ewLo[r] : ewHi[r - 4];
      const int row = r + 8 * half;            // C layout: VGPR r -> M = r + 8*half
      my[row][l15] = gnn_sigmoid(c1a[r] + w * w1ewA + b1a);
      my[row][cb]  = gnn_sigmoid(c1b[r] + w * w1ewB + b1b);
    }
    __syncthreads();

    // Re-read as A fragment for layer 2 (row per lane, ISA half/K interleave).
    v16h a2;
    const float* hrow = my[l15];
#pragma unroll
    for (int j = 0; j < 8; ++j) {
      a2[j]     = (_Float16)hrow[8 * half + j];
      a2[8 + j] = (_Float16)hrow[16 + 8 * half + j];
    }

    v8f c2a = {};
    v8f c2b = {};
    c2a = __builtin_amdgcn_wmma_f32_16x16x32_f16(false, a2, false, B2a, (short)0, c2a, false, false);
    c2b = __builtin_amdgcn_wmma_f32_16x16x32_f16(false, a2, false, B2b, (short)0, c2b, false, false);

    // Signed scatter: +h into tgt row, -h into src row. Lanes cover consecutive
    // columns of one node row -> coalesced L2 atomics. Ids broadcast via shfl
    // (lane m holds ids of edge e0+m for m<16).
#pragma unroll
    for (int r = 0; r < 8; ++r) {
      const int row = r + 8 * half;
      const int tn = __shfl(t0, row, 32);
      const int sn = __shfl(s0, row, 32);
      const float ha = gnn_sigmoid(c2a[r] + b2a);
      atomicAdd(&aggb[(size_t)tn * kAggStride + l15],  ha);
      atomicAdd(&aggb[(size_t)sn * kAggStride + l15], -ha);
      if (cb < kEOUT) {
        const float hb = gnn_sigmoid(c2b[r] + b2b);
        atomicAdd(&aggb[(size_t)tn * kAggStride + cb],  hb);
        atomicAdd(&aggb[(size_t)sn * kAggStride + cb], -hb);
      }
    }
    __syncthreads();
  }
}

// ---------------------------------------------------------------------------
// Node MLP: out = sigmoid(agg @ W3 + b3). One wave = 16 nodes, one WMMA.
// ---------------------------------------------------------------------------
__global__ __launch_bounds__(256) void gnn_node_kernel(
    const float* __restrict__ agg, const float* __restrict__ W3,
    const float* __restrict__ b3v, float* __restrict__ out)
{
  const int lane = threadIdx.x & 31;
  const int wid  = threadIdx.x >> 5;
  const int half = lane >> 4;
  const int l15  = lane & 15;
  const int koff = half * 16;
  const int b    = blockIdx.y;
  const int tile = blockIdx.x * 8 + wid;
  if (tile >= kNodeTiles) return;            // whole-wave uniform exit (EXEC stays all-1)

  v16h B3;
#pragma unroll
  for (int j = 0; j < 16; ++j) {
    const int k = (j < 8) ? 8 * half + j : 16 + 8 * half + (j - 8);
    B3[j] = (_Float16)((k < kEOUT) ? W3[k * kNOUT + l15] : 0.0f);
  }
  const float bias = b3v[l15];

  const float* aggb = agg + (size_t)b * kN * kAggStride;
  float*       outb = out + (size_t)b * kN * kNOUT;

  const int n0 = tile * 16;
  const float* arow = aggb + (size_t)(n0 + l15) * kAggStride;
  const v4f a0 = *(const v4f*)(arow + 8 * half);          // K = 8h .. 8h+3
  const v4f a1 = *(const v4f*)(arow + 8 * half + 4);      // K = 8h+4 .. 8h+7
  const v4f a2 = *(const v4f*)(arow + 16 + 8 * half);     // K = 16+8h ..
  const v4f a3 = *(const v4f*)(arow + 16 + 8 * half + 4); // cols 30,31 are zero

  v16h a;
#pragma unroll
  for (int j = 0; j < 4; ++j) {
    a[j]      = (_Float16)a0[j];
    a[4 + j]  = (_Float16)a1[j];
    a[8 + j]  = (_Float16)a2[j];
    a[12 + j] = (_Float16)a3[j];
  }

  v8f c = {};
  c = __builtin_amdgcn_wmma_f32_16x16x32_f16(false, a, false, B3, (short)0, c, false, false);

#pragma unroll
  for (int r = 0; r < 8; ++r) {
    const int row = r + 8 * half;
    outb[(size_t)(n0 + row) * kNOUT + l15] = gnn_sigmoid(c[r] + bias);
  }
}

// ---------------------------------------------------------------------------
extern "C" void kernel_launch(void* const* d_in, const int* in_sizes, int n_in,
                              void* d_out, int out_size, void* d_ws, size_t ws_size,
                              hipStream_t stream) {
  (void)in_sizes; (void)n_in; (void)out_size; (void)ws_size;
  const float* nf  = (const float*)d_in[0];   // node_features (B,N,F)
  const float* ew  = (const float*)d_in[1];   // edge_weight   (B,E)
  const int*   idx = (const int*)d_in[2];     // edge_index    (2,E)
  const float* W1  = (const float*)d_in[3];   // (33,32)
  const float* b1  = (const float*)d_in[4];   // (32,)
  const float* W2  = (const float*)d_in[5];   // (32,30)
  const float* b2  = (const float*)d_in[6];   // (30,)
  const float* W3  = (const float*)d_in[7];   // (30,16)
  const float* b3  = (const float*)d_in[8];   // (16,)
  float* out = (float*)d_out;                 // (B,N,16)
  float* agg = (float*)d_ws;                  // (B,N,32) padded accumulator

  const int n4 = kB * kN * kAggStride / 4;
  gnn_zero_kernel<<<512, 256, 0, stream>>>((float4*)agg, n4);
  gnn_edge_kernel<<<dim3(kEdgeBlocks, kB), 256, 0, stream>>>(nf, ew, idx, W1, b1, W2, b2, agg);
  gnn_node_kernel<<<dim3(kNodeBlocks, kB), 256, 0, stream>>>(agg, W3, b3, out);
}
